// Cross_Attention_bilateral_1632087572978
// MI455X (gfx1250) — matrix operand order repaired and checked by reading, not compiled, not run on hardware
//
#include <hip/hip_runtime.h>

// ---------------------------------------------------------------------------
// Bilateral cross-attention for MI455X (gfx1250, wave32, WMMA).
//   1) proj_kernel      : 1x1 convs -> bf16 Q/K [b][pos][32], V [b][32][pos].
//   2) attn_kernel      : fused flash attention; v_wmma_f32_16x16x32_bf16 for
//                         S = Q*K^T, O += P*V, AND row-sums (P x ones trick);
//                         online softmax max via half-wave shfl_xor.
//   3) prep_M/prep_cb   : fold wo[:, half] @ wv into 256x32 matrices.
//   4) epilogue_kernel  : out = x + M_a @ O_a + M_b @ O_b + cb (residual).
// ---------------------------------------------------------------------------

#define BSZ 2
#define CDIM 256
#define CHD 32
#define NPOS 4096  // 64*64

typedef __bf16 v16bf __attribute__((ext_vector_type(16)));
typedef float  v8f   __attribute__((ext_vector_type(8)));
typedef unsigned int u32x4 __attribute__((ext_vector_type(4)));

union BF16Frag { u32x4 u[2]; v16bf v; };

__device__ __forceinline__ unsigned short f2bf(float f) {
  unsigned int u = __float_as_uint(f);
  u += 0x7FFFu + ((u >> 16) & 1u);  // round-to-nearest-even
  return (unsigned short)(u >> 16);
}

// ---------------------------------------------------------------------------
// Projection: out[ch, pos] = sum_c W[ch, c] * in[b, c, pos] + bias[ch]
// vmode 0: bf16 [b][pos][32] (Q/K rows); vmode 1: bf16 [b][32][pos] (V^T)
// ---------------------------------------------------------------------------
__global__ __launch_bounds__(256) void proj_kernel(
    const float* __restrict__ in, const float* __restrict__ W,
    const float* __restrict__ bias, unsigned short* __restrict__ out,
    int vmode) {
  int idx = blockIdx.x * 256 + threadIdx.x;   // over BSZ*NPOS = 8192
  int b = idx >> 12;
  int p = idx & (NPOS - 1);
  float acc[CHD];
#pragma unroll
  for (int ch = 0; ch < CHD; ++ch) acc[ch] = bias[ch];
  const float* xp = in + (size_t)b * CDIM * NPOS + p;
  for (int c = 0; c < CDIM; ++c) {
    float xv = xp[(size_t)c * NPOS];           // coalesced across lanes
#pragma unroll
    for (int ch = 0; ch < CHD; ++ch)           // W reads wave-uniform -> s_load
      acc[ch] = fmaf(W[ch * CDIM + c], xv, acc[ch]);
  }
  if (vmode == 0) {
    unsigned int words[16];
#pragma unroll
    for (int k = 0; k < 16; ++k)
      words[k] = (unsigned)f2bf(acc[2 * k]) | ((unsigned)f2bf(acc[2 * k + 1]) << 16);
    u32x4* o4 = (u32x4*)(out + (size_t)idx * CHD);
#pragma unroll
    for (int i = 0; i < 4; ++i) {
      u32x4 t;
      t[0] = words[4 * i]; t[1] = words[4 * i + 1];
      t[2] = words[4 * i + 2]; t[3] = words[4 * i + 3];
      o4[i] = t;                               // global_store_b128
    }
  } else {
#pragma unroll
    for (int ch = 0; ch < CHD; ++ch)
      out[((size_t)(b * CHD + ch)) * NPOS + p] = f2bf(acc[ch]);
  }
}

// ---------------------------------------------------------------------------
// Fused flash attention. One wave = one 16-query block of one (batch, combo).
// combo: 0=(Q1,K1,V1) 1=(Q1,K2,V2) 2=(Q2,K1,V1) 3=(Q2,K2,V2)
// Tasks = 2*4*256 = 2048 full waves -> grid 256 x 256 threads.
// ---------------------------------------------------------------------------
__global__ __launch_bounds__(256) void attn_kernel(
    const unsigned short* __restrict__ Q1, const unsigned short* __restrict__ Q2,
    const unsigned short* __restrict__ K1, const unsigned short* __restrict__ K2,
    const unsigned short* __restrict__ V1, const unsigned short* __restrict__ V2,
    float* __restrict__ O) {
  // per-wave P tile: 16 rows x 32 cols bf16, row stride 40 halves (16B aligned)
  __shared__ __align__(16) unsigned short plds[8][16 * 40];

  const int lane = threadIdx.x & 31;
  const int w    = threadIdx.x >> 5;
  const int task = blockIdx.x * 8 + w;
  const int qblk  = task & 255;
  const int combo = (task >> 8) & 3;
  const int b     = (task >> 10) & 1;

  const unsigned short* Q = (combo & 2) ? Q2 : Q1;
  const unsigned short* K = (combo & 1) ? K2 : K1;
  const unsigned short* V = (combo & 1) ? V2 : V1;

  const int hs = lane >> 4;   // half-wave select
  const int ln = lane & 15;

  // ---- Q A-fragment (16x32 bf16): lane row = ln, contiguous k = 16*hs + s ----
  const char* qbase = (const char*)Q + (size_t)(b * NPOS + qblk * 16 + ln) * 64;
  BF16Frag qf;
  qf.u[0] = *(const u32x4*)(qbase + hs * 32);        // K = hs*16 .. +7
  qf.u[1] = *(const u32x4*)(qbase + hs * 32 + 16);   // K = hs*16+8 .. +15

  // ones B-fragment (bf16 1.0 everywhere) for tensor-core row sums
  BF16Frag onef;
  {
    u32x4 o1s;
    o1s[0] = 0x3F803F80u; o1s[1] = 0x3F803F80u;
    o1s[2] = 0x3F803F80u; o1s[3] = 0x3F803F80u;
    onef.u[0] = o1s; onef.u[1] = o1s;
  }

  v8f o0 = {}, o1 = {};
  v8f mi, li;
#pragma unroll
  for (int r = 0; r < 8; ++r) { mi[r] = -3.0e38f; li[r] = 0.0f; }

  const char* kbb = (const char*)K + (size_t)b * NPOS * 64;
  const char* vbb = (const char*)V + (size_t)b * CHD * NPOS * 2;
  unsigned short* pw = plds[w];

#pragma unroll 2
  for (int kt = 0; kt < NPOS / 32; ++kt) {
    // ---- K B-fragments (32ch x 16keys): lane col = key, 16 contig ch halves
    const char* ka = kbb + (size_t)(kt * 32 + ln) * 64 + hs * 32;
    const char* kb = ka + 16 * 64;
    BF16Frag kf0, kf1;
    kf0.u[0] = *(const u32x4*)ka;       kf0.u[1] = *(const u32x4*)(ka + 16);
    kf1.u[0] = *(const u32x4*)kb;       kf1.u[1] = *(const u32x4*)(kb + 16);
    __builtin_prefetch(ka + 2048, 0, 3);   // next K tile, keep near

    // ---- V B-fragments (32keys x 16d): contiguous keys (transposed layout)
    const char* va = vbb + (size_t)ln * (NPOS * 2) + (size_t)kt * 64 + hs * 32;
    const char* vb = va + (size_t)16 * (NPOS * 2);
    BF16Frag vf0, vf1;
    vf0.u[0] = *(const u32x4*)va;       vf0.u[1] = *(const u32x4*)(va + 16);
    vf1.u[0] = *(const u32x4*)vb;       vf1.u[1] = *(const u32x4*)(vb + 16);

    // ---- S = Q * K^T  (two 16x16 f32 tiles covering 32 keys) ----
    v8f z = {};
    v8f s0 = __builtin_amdgcn_wmma_f32_16x16x32_bf16(false, qf.v, false, kf0.v,
                                                     (short)0, z, false, false);
    v8f s1 = __builtin_amdgcn_wmma_f32_16x16x32_bf16(false, qf.v, false, kf1.v,
                                                     (short)0, z, false, false);

    // ---- row max: row m = r + 8*hs lives across a 16-lane half-wave ----
    v8f tm;
#pragma unroll
    for (int r = 0; r < 8; ++r) tm[r] = fmaxf(s0[r], s1[r]);
#pragma unroll
    for (int msk = 1; msk < 16; msk <<= 1)
#pragma unroll
      for (int r = 0; r < 8; ++r) tm[r] = fmaxf(tm[r], __shfl_xor(tm[r], msk, 32));

    v8f p0, p1, sc;
#pragma unroll
    for (int r = 0; r < 8; ++r) {
      float mn = fmaxf(mi[r], tm[r]);
      sc[r] = __expf(mi[r] - mn);
      p0[r] = __expf(s0[r] - mn);
      p1[r] = __expf(s1[r] - mn);
      mi[r] = mn;
    }

    // ---- P: C-fragment -> bf16 A-fragment via per-wave LDS tile ----
#pragma unroll
    for (int r = 0; r < 8; ++r) {
      int row = r + hs * 8;
      pw[row * 40 + ln]      = f2bf(p0[r]);
      pw[row * 40 + 16 + ln] = f2bf(p1[r]);
    }
    // same-wave DS ops are in-order; compiler inserts s_wait_dscnt for RAW
    BF16Frag pf;
    const unsigned short* pr = pw + ln * 40;
    pf.u[0] = *(const u32x4*)(pr + hs * 16);       // keys hs*16..+7
    pf.u[1] = *(const u32x4*)(pr + hs * 16 + 8);   // keys hs*16+8..+15

    // ---- row sums on the tensor core: rs[m] = sum_k P[m,k] * 1 ----
    v8f rsf = __builtin_amdgcn_wmma_f32_16x16x32_bf16(false, pf.v, false, onef.v,
                                                      (short)0, z, false, false);

    // ---- rescale state + accumulators, then O += P * V ----
#pragma unroll
    for (int r = 0; r < 8; ++r) {
      li[r] = li[r] * sc[r] + rsf[r];
      o0[r] *= sc[r];
      o1[r] *= sc[r];
    }
    o0 = __builtin_amdgcn_wmma_f32_16x16x32_bf16(false, pf.v, false, vf0.v,
                                                 (short)0, o0, false, false);
    o1 = __builtin_amdgcn_wmma_f32_16x16x32_bf16(false, pf.v, false, vf1.v,
                                                 (short)0, o1, false, false);
  }

  // ---- normalize and store O as f32 [combo][b][pos][32] ----
  float* Op = O + (size_t)combo * (BSZ * NPOS * CHD) +
              ((size_t)b * NPOS + qblk * 16) * CHD;
#pragma unroll
  for (int r = 0; r < 8; ++r) {
    float inv = 1.0f / li[r];
    int row = r + hs * 8;
    Op[row * CHD + ln]      = o0[r] * inv;
    Op[row * CHD + 16 + ln] = o1[r] * inv;
  }
}

// ---------------------------------------------------------------------------
// M[o][ch] = sum_c wo[o][halfsel*256 + c] * wv[c][ch]   (256x32)
// ---------------------------------------------------------------------------
__global__ __launch_bounds__(256) void prep_M_kernel(
    const float* __restrict__ wo, const float* __restrict__ wv,
    float* __restrict__ M, int halfsel) {
  int t = blockIdx.x * 256 + threadIdx.x;   // 8192
  int o = t >> 5, ch = t & 31;
  const float* wrow = wo + (size_t)o * 512 + halfsel * 256;
  float acc = 0.0f;
  for (int c = 0; c < 256; ++c) acc = fmaf(wrow[c], wv[(size_t)c * 32 + ch], acc);
  M[t] = acc;
}

__global__ __launch_bounds__(256) void prep_cb_kernel(
    const float* __restrict__ wo, const float* __restrict__ bva,
    const float* __restrict__ bvb, const float* __restrict__ bo,
    float* __restrict__ cb) {
  int o = threadIdx.x;   // 256 threads, 1 block
  float acc = bo[o];
  for (int c = 0; c < 256; ++c)
    acc += wo[(size_t)o * 512 + c] * bva[c] + wo[(size_t)o * 512 + 256 + c] * bvb[c];
  cb[o] = acc;
}

// ---------------------------------------------------------------------------
// out[b][o][p] = x[b][o][p] + cb[o] + sum_ch Ma[o][ch]*Oa[p][ch] + Mb[o][ch]*Ob[p][ch]
// ---------------------------------------------------------------------------
__global__ __launch_bounds__(256) void epilogue_kernel(
    const float* __restrict__ xin, const float* __restrict__ Oa,
    const float* __restrict__ Ob, const float* __restrict__ Ma,
    const float* __restrict__ Mb, const float* __restrict__ cb,
    float* __restrict__ out) {
  int idx = blockIdx.x * 256 + threadIdx.x;   // BSZ*NPOS
  int b = idx >> 12, p = idx & (NPOS - 1);
  float oa[CHD], ob[CHD];
  const float* Oar = Oa + (size_t)idx * CHD;
  const float* Obr = Ob + (size_t)idx * CHD;
#pragma unroll
  for (int i = 0; i < CHD; ++i) { oa[i] = Oar[i]; ob[i] = Obr[i]; }
  const float* xb = xin + (size_t)b * CDIM * NPOS + p;
  float* op = out + (size_t)b * CDIM * NPOS + p;
  for (int o = 0; o < CDIM; ++o) {
    float acc = cb[o];                       // uniform -> scalar loads
    const float* ma = Ma + o * CHD;
    const float* mb = Mb + o * CHD;
#pragma unroll
    for (int ch = 0; ch < CHD; ++ch)
      acc = fmaf(ma[ch], oa[ch], fmaf(mb[ch], ob[ch], acc));
    op[(size_t)o * NPOS] = xb[(size_t)o * NPOS] + acc;
  }
}

// ---------------------------------------------------------------------------
extern "C" void kernel_launch(void* const* d_in, const int* in_sizes, int n_in,
                              void* d_out, int out_size, void* d_ws, size_t ws_size,
                              hipStream_t stream) {
  const float* x    = (const float*)d_in[0];
  const float* y    = (const float*)d_in[1];
  const float* wf1  = (const float*)d_in[2];   const float* bf1 = (const float*)d_in[3];
  const float* wg1  = (const float*)d_in[4];   const float* bg1 = (const float*)d_in[5];
  const float* wh1  = (const float*)d_in[6];   const float* bh1 = (const float*)d_in[7];
  const float* wf2  = (const float*)d_in[8];   const float* bf2 = (const float*)d_in[9];
  const float* wg2  = (const float*)d_in[10];  const float* bg2 = (const float*)d_in[11];
  const float* wh2  = (const float*)d_in[12];  const float* bh2 = (const float*)d_in[13];
  const float* wv11 = (const float*)d_in[14];  const float* bv11 = (const float*)d_in[15];
  const float* wv12 = (const float*)d_in[16];  const float* bv12 = (const float*)d_in[17];
  const float* wv21 = (const float*)d_in[18];  const float* bv21 = (const float*)d_in[19];
  const float* wv22 = (const float*)d_in[20];  const float* bv22 = (const float*)d_in[21];
  const float* wo1  = (const float*)d_in[22];  const float* bo1 = (const float*)d_in[23];
  const float* wo2  = (const float*)d_in[24];  const float* bo2 = (const float*)d_in[25];

  char* ws = (char*)d_ws;
  const size_t BN = (size_t)BSZ * NPOS;          // 8192
  const size_t bf16_arr = BN * CHD * 2;          // 512 KB each
  unsigned short* Q1 = (unsigned short*)(ws + 0 * bf16_arr);
  unsigned short* Q2 = (unsigned short*)(ws + 1 * bf16_arr);
  unsigned short* K1 = (unsigned short*)(ws + 2 * bf16_arr);
  unsigned short* K2 = (unsigned short*)(ws + 3 * bf16_arr);
  unsigned short* V1 = (unsigned short*)(ws + 4 * bf16_arr);
  unsigned short* V2 = (unsigned short*)(ws + 5 * bf16_arr);
  float* O   = (float*)(ws + 6 * bf16_arr);      // 4 combos * BN * 32 f32 = 4 MB
  float* M11 = (float*)(ws + 6 * bf16_arr + 4 * BN * CHD * 4);
  float* M12 = M11 + 8192;
  float* M21 = M12 + 8192;
  float* M22 = M21 + 8192;
  float* cb1 = M22 + 8192;
  float* cb2 = cb1 + 256;

  // 1) projections (h2 comes from x, matching the reference)
  proj_kernel<<<32, 256, 0, stream>>>(x, wf1, bf1, Q1, 0);
  proj_kernel<<<32, 256, 0, stream>>>(y, wf2, bf2, Q2, 0);
  proj_kernel<<<32, 256, 0, stream>>>(x, wg1, bg1, K1, 0);
  proj_kernel<<<32, 256, 0, stream>>>(y, wg2, bg2, K2, 0);
  proj_kernel<<<32, 256, 0, stream>>>(x, wh1, bh1, V1, 1);
  proj_kernel<<<32, 256, 0, stream>>>(x, wh2, bh2, V2, 1);

  // 2) fold output projections
  prep_M_kernel<<<32, 256, 0, stream>>>(wo1, wv11, M11, 0);
  prep_M_kernel<<<32, 256, 0, stream>>>(wo1, wv12, M12, 1);
  prep_M_kernel<<<32, 256, 0, stream>>>(wo2, wv21, M21, 0);
  prep_M_kernel<<<32, 256, 0, stream>>>(wo2, wv22, M22, 1);
  prep_cb_kernel<<<1, 256, 0, stream>>>(wo1, bv11, bv12, bo1, cb1);
  prep_cb_kernel<<<1, 256, 0, stream>>>(wo2, bv21, bv22, bo2, cb2);

  // 3) fused flash attention, all 8 instances
  attn_kernel<<<256, 256, 0, stream>>>(Q1, Q2, K1, K2, V1, V2, O);

  // 4) epilogue with residual
  float* O11 = O;
  float* O12 = O + 1 * BN * CHD;
  float* O21 = O + 2 * BN * CHD;
  float* O22 = O + 3 * BN * CHD;
  float* out = (float*)d_out;
  epilogue_kernel<<<32, 256, 0, stream>>>(x, O11, O12, M11, M12, cb1, out);
  epilogue_kernel<<<32, 256, 0, stream>>>(y, O21, O22, M21, M22, cb2,
                                          out + (size_t)BSZ * CDIM * NPOS);
}
